// DHMSA_41592463294756
// MI455X (gfx1250) — matrix operand order repaired and checked by reading, not compiled, not run on hardware
//
#include <hip/hip_runtime.h>
#include <hip/hip_bf16.h>

// ---------------- problem constants ----------------
constexpr int BN    = 2;
constexpr int HIM   = 126;
constexpr int WIM   = 126;
constexpr int C     = 256;
constexpr int HP    = 128;           // padded H/W
constexpr int WP    = 128;
constexpr int NPIX  = BN * HP * WP;  // 32768 padded pixels
constexpr int CW    = 8;
constexpr int HWIN  = 16;
constexpr int WL    = 64;            // queries per window
constexpr int HL    = 256;           // halo keys per window
constexpr int HEADS = 8;
constexpr int HD    = 32;
constexpr int NWIN  = BN * 16 * 16;  // 512 windows
constexpr int C3    = 3 * C;         // 768

typedef _Float16 h16v __attribute__((ext_vector_type(16)));
typedef float    f8v  __attribute__((ext_vector_type(8)));

union FragH { h16v h; uint4 u4[2]; };

// A-fragment (16x32, f16), row-major source, ldh = row stride in halves.
// lane L: row = L&15;  element e -> k = (e>>3)*16 + (L>>4)*8 + (e&7)
__device__ __forceinline__ h16v load_a_rm(const _Float16* base, int ldh, int lane) {
    int row = lane & 15, g = lane >> 4;
    FragH f;
    f.u4[0] = *reinterpret_cast<const uint4*>(base + (size_t)row * ldh + g * 8);
    f.u4[1] = *reinterpret_cast<const uint4*>(base + (size_t)row * ldh + 16 + g * 8);
    return f.h;
}

// B-fragment (32x16, f16) from N-major storage [n][k], ldh = k-stride per n row.
// lane L: n = L&15; element e -> k = (e>>3)*16 + (L>>4)*8 + (e&7)
__device__ __forceinline__ h16v load_b_nm(const _Float16* base, int ldh, int lane) {
    int n = lane & 15, g = lane >> 4;
    const _Float16* p = base + (size_t)n * ldh + g * 8;
    FragH f;
    f.u4[0] = *reinterpret_cast<const uint4*>(p);
    f.u4[1] = *reinterpret_cast<const uint4*>(p + 16);
    return f.h;
}

__device__ __forceinline__ f8v wmma16(h16v a, h16v b, f8v c) {
    return __builtin_amdgcn_wmma_f32_16x16x32_f16(false, a, false, b, (short)0, c, false, false);
}

// Async global->LDS copy of 16 bytes per lane (GVS mode: SGPR base + VGPR offset).
__device__ __forceinline__ void async_cp16(uint32_t lds_addr, uint32_t goff, uint64_t gbase) {
    asm volatile("global_load_async_to_lds_b128 %0, %1, %2"
                 :: "v"(lds_addr), "v"(goff), "s"(gbase)
                 : "memory");
}
__device__ __forceinline__ void wait_async0() {
#if __has_builtin(__builtin_amdgcn_s_wait_asynccnt)
    __builtin_amdgcn_s_wait_asynccnt(0);
#else
    asm volatile("s_wait_asynccnt 0" ::: "memory");
#endif
}

// ---------------- kernel 1: weight prep (f32 -> f16, N-major) ----------------
__global__ void k_prep(const float* __restrict__ w_qkv, const float* __restrict__ proj_w,
                       _Float16* __restrict__ wqkvT, _Float16* __restrict__ projT) {
    int tid = blockIdx.x * blockDim.x + threadIdx.x;
    if (tid < C3 * C) {                       // wqkvT[n][k] = w_qkv[k][n]
        int n = tid / C, k = tid % C;
        wqkvT[tid] = (_Float16)w_qkv[(size_t)k * C3 + n];
    } else if (tid < C3 * C + C * C) {        // projT[n][k] = proj_w[k][n]
        int t = tid - C3 * C;
        int n = t / C, k = t % C;
        projT[t] = (_Float16)proj_w[(size_t)k * C + n];
    }
}

// ---------------- kernel 2: continuous-position-bias table tab[529][8] ----------------
__global__ void k_cpb(const float* __restrict__ w0, const float* __restrict__ b0,
                      const float* __restrict__ w1, float* __restrict__ tab) {
    int tid = blockIdx.x * blockDim.x + threadIdx.x;
    if (tid >= 529 * 8) return;
    int head = tid & 7, idx = tid >> 3;
    int ri = idx / 23, rj = idx % 23;
    auto relf = [](int t) {
        float x = (float)(t - 11) * (8.0f / 7.0f);
        float s = (x > 0.f) ? 1.f : ((x < 0.f) ? -1.f : 0.f);
        return s * log1pf(fabsf(x)) / logf(8.0f);
    };
    float v0 = relf(ri), v1 = relf(rj);
    float acc = 0.f;
    for (int j = 0; j < 512; j++) {
        float hj = fmaxf(v0 * w0[j] + v1 * w0[512 + j] + b0[j], 0.f);
        acc += hj * w1[j * 8 + head];
    }
    tab[idx * 8 + head] = 16.f / (1.f + expf(-acc));
}

// ---------------- kernel 3: depthwise conv 3x3 + LayerNorm -> y f16 (padded grid) ----------------
__global__ void k_conv_ln(const float* __restrict__ x, const float* __restrict__ dwk,
                          const float* __restrict__ gamma, const float* __restrict__ beta,
                          _Float16* __restrict__ y) {
    int pix = blockIdx.x;              // over BN*HP*WP
    int c = threadIdx.x;               // 256 channels
    int w = pix & 127, h = (pix >> 7) & 127, b = pix >> 14;
    bool valid = (h < HIM) && (w < WIM);
    __shared__ float red[256];
    float v = 0.f;
    if (valid) {
        for (int dh = 0; dh < 3; dh++) {
            int hh = h + dh - 1;
            if (hh < 0 || hh >= HIM) continue;
            for (int dw = 0; dw < 3; dw++) {
                int ww = w + dw - 1;
                if (ww < 0 || ww >= WIM) continue;
                v += x[(((size_t)b * HIM + hh) * WIM + ww) * C + c] * dwk[(dh * 3 + dw) * C + c];
            }
        }
    }
    red[c] = v; __syncthreads();
    for (int s = 128; s > 0; s >>= 1) { if (c < s) red[c] += red[c + s]; __syncthreads(); }
    float mu = red[0] * (1.f / 256.f); __syncthreads();
    float d = v - mu;
    red[c] = d * d; __syncthreads();
    for (int s = 128; s > 0; s >>= 1) { if (c < s) red[c] += red[c + s]; __syncthreads(); }
    float var = red[0] * (1.f / 256.f);
    float out = valid ? (d * rsqrtf(var + 1e-5f) * gamma[c] + beta[c]) : 0.f;
    y[(size_t)pix * C + c] = (_Float16)out;
}

// ---------------- kernel 4: QKV GEMM  qkv[32768 x 768] = y @ w_qkv + bias ----------------
__global__ __launch_bounds__(128) void k_qkv(const _Float16* __restrict__ y,
                                             const _Float16* __restrict__ wT,
                                             const float* __restrict__ q_bias,
                                             const float* __restrict__ v_bias,
                                             _Float16* __restrict__ qkv) {
    int blk = blockIdx.x;                  // 512 m-blocks * 6 n-blocks
    int bn = blk % 6, bm = blk / 6;
    int lane = threadIdx.x & 31, wv = threadIdx.x >> 5;
    int m0 = bm * 64 + wv * 16;
    int n0 = bn * 128;
    f8v acc[8] = {};
    for (int ks = 0; ks < 8; ks++) {
        h16v a = load_a_rm(y + (size_t)m0 * C + ks * 32, C, lane);
        for (int t = 0; t < 8; t++) {
            h16v b = load_b_nm(wT + (size_t)(n0 + t * 16) * C + ks * 32, C, lane);
            acc[t] = wmma16(a, b, acc[t]);
        }
    }
    int g = lane >> 4, nlo = lane & 15;
    for (int t = 0; t < 8; t++) {
        int n = n0 + t * 16 + nlo;
        float bias = (n < C) ? q_bias[n] : ((n < 2 * C) ? 0.f : v_bias[n - 2 * C]);
        for (int r = 0; r < 8; r++) {
            int row = m0 + r + 8 * g;
            int w = row & 127, h = (row >> 7) & 127;
            float v = acc[t][r];
            v = (h < HIM && w < WIM) ? (v + bias) : 0.f;   // pad-after-bias => 0 in pad region
            qkv[(size_t)row * C3 + n] = (_Float16)v;
        }
    }
}

// ---------------- kernel 5: window/halo gather + L2-norm; vT transpose ----------------
__global__ void k_gather(const _Float16* __restrict__ qkv, const float* __restrict__ scale_logit,
                         _Float16* __restrict__ qL2, _Float16* __restrict__ kL2,
                         _Float16* __restrict__ vT) {
    int tid = blockIdx.x * blockDim.x + threadIdx.x;   // NWIN*HL*HEADS
    int head = tid & 7, pos = (tid >> 3) & 255, win = tid >> 11;
    if (win >= NWIN) return;
    int wj = win & 15, wi = (win >> 4) & 15, b = win >> 8;
    size_t wh = (size_t)win * HEADS + head;

    // K / V from halo window
    int r = wi * 8 - 4 + (pos >> 4);
    int cc = wj * 8 - 4 + (pos & 15);
    float kv[HD], vv[HD], ss = 0.f;
    if (r >= 0 && r < HP && cc >= 0 && cc < WP) {
        size_t ro = ((size_t)((b * HP + r) * WP + cc)) * C3;
        const _Float16* kp = qkv + ro + C + head * HD;
        const _Float16* vp = qkv + ro + 2 * C + head * HD;
        for (int d = 0; d < HD; d++) { kv[d] = (float)kp[d]; ss += kv[d] * kv[d]; vv[d] = (float)vp[d]; }
    } else {
        for (int d = 0; d < HD; d++) { kv[d] = 0.f; vv[d] = 0.f; }
    }
    float rs = rsqrtf(fmaxf(ss, 1e-12f));
    size_t kb = (wh * HL + pos) * HD;
    for (int d = 0; d < HD; d++) kL2[kb + d] = (_Float16)(kv[d] * rs);
    size_t vb = (wh * HD) * HL + pos;                  // vT[win][head][d][pos]
    for (int d = 0; d < HD; d++) vT[vb + (size_t)d * HL] = (_Float16)vv[d];

    // Q (only first 64 positions)
    if (pos < WL) {
        int hq = wi * 8 + (pos >> 3), wq = wj * 8 + (pos & 7);
        size_t ro = ((size_t)((b * HP + hq) * WP + wq)) * C3;
        const _Float16* qp = qkv + ro + head * HD;
        float qv[HD], qs = 0.f;
        for (int d = 0; d < HD; d++) { qv[d] = (float)qp[d]; qs += qv[d] * qv[d]; }
        float sc = rsqrtf(fmaxf(qs, 1e-12f)) * expf(scale_logit[head]);
        size_t qb = (wh * WL + pos) * HD;
        for (int d = 0; d < HD; d++) qL2[qb + d] = (_Float16)(qv[d] * sc);
    }
}

// ---------------- kernel 6: attention per (window, head) ----------------
// block = 128 threads (4 waves); wave = one 16-row M-tile of the 64 queries.
// K (16KB) and V (16KB) tiles are staged once per block into LDS with async
// global->LDS copies (ASYNCcnt); K tile aliases the first 16KB of the P buffer.
__global__ __launch_bounds__(128) void k_attn(const _Float16* __restrict__ qL2,
                                              const _Float16* __restrict__ kL2,
                                              const _Float16* __restrict__ vT,
                                              const float* __restrict__ tab,
                                              _Float16* __restrict__ attnout) {
    constexpr int LDP = 264;                            // padded P row (halves)
    __shared__ _Float16 ldsP[4][16][LDP];               // 33792 B (first 16KB doubles as K tile)
    __shared__ _Float16 ldsV[HD][HL];                   // 16384 B  [hd][pos]
    int win = blockIdx.x >> 3, head = blockIdx.x & 7;
    int lane = threadIdx.x & 31, wv = threadIdx.x >> 5;
    int wj = win & 15, wi = (win >> 4) & 15;
    size_t wh = (size_t)win * HEADS + head;

    // ---- async stage: K tile (contiguous 16KB) and V tile (contiguous 16KB) ----
    const _Float16* kTile = &ldsP[0][0][0];             // [pos][hd], byte-for-byte copy of kL2 tile
    uint64_t kbase = (uint64_t)(uintptr_t)(kL2 + wh * HL * HD);
    uint64_t vbase = (uint64_t)(uintptr_t)(vT + wh * HD * HL);
    uint32_t ldsK0 = (uint32_t)(uintptr_t)&ldsP[0][0][0];   // low 32 bits == LDS byte address
    uint32_t ldsV0 = (uint32_t)(uintptr_t)&ldsV[0][0];
    for (int i = 0; i < 8; i++) {
        uint32_t off = (uint32_t)(threadIdx.x + i * 128) * 16;   // 128 thr * 16B * 8 = 16KB
        async_cp16(ldsK0 + off, off, kbase);
        async_cp16(ldsV0 + off, off, vbase);
    }
    wait_async0();
    __syncthreads();

    // ---- S = q(16x32) @ k^T(32x256) : 16 WMMAs, A reused, B from LDS ----
    h16v aq = load_a_rm(qL2 + (wh * WL + wv * 16) * HD, HD, lane);
    f8v acc[16] = {};
    for (int t = 0; t < 16; t++) {
        h16v bk = load_b_nm(kTile + (size_t)(t * 16) * HD, HD, lane);
        acc[t] = wmma16(aq, bk, acc[t]);
    }
    __syncthreads();                                    // all K reads done before P overwrites

    int g = lane >> 4, nlo = lane & 15;
    // ---- bias + mask + row softmax (rows live across 16-lane groups) ----
    for (int r = 0; r < 8; r++) {
        int m = wv * 16 + r + 8 * g;                    // query position 0..63
        int qi = m >> 3, qj = m & 7;
        float sv[16], mx = -1e30f;
        for (int t = 0; t < 16; t++) {
            int kpos = t * 16 + nlo;
            int ki = kpos >> 4, kj = kpos & 15;
            int idx = (qi - ki + 15) * 23 + (qj - kj + 15);
            float bias = tab[idx * 8 + head];
            int rr = wi * 8 - 4 + ki, ccc = wj * 8 - 4 + kj;
            float msk = (rr >= 0 && rr < HIM && ccc >= 0 && ccc < WIM) ? 0.f : -100.f;
            float s = acc[t][r] + bias + msk;
            sv[t] = s;
            mx = fmaxf(mx, s);
        }
        for (int o = 1; o < 16; o <<= 1) mx = fmaxf(mx, __shfl_xor(mx, o, 32));
        float sm = 0.f;
        for (int t = 0; t < 16; t++) { sv[t] = __expf(sv[t] - mx); sm += sv[t]; }
        for (int o = 1; o < 16; o <<= 1) sm += __shfl_xor(sm, o, 32);
        float inv = 1.f / sm;
        for (int t = 0; t < 16; t++)
            ldsP[wv][r + 8 * g][t * 16 + nlo] = (_Float16)(sv[t] * inv);
    }
    __syncthreads();

    // ---- O = P(16x256) @ V(256x32) : 2 N-tiles x 8 K-steps, both operands from LDS ----
    for (int no = 0; no < 2; no++) {
        f8v ao = {};
        for (int ks = 0; ks < 8; ks++) {
            h16v ap = load_a_rm(&ldsP[wv][0][ks * 32], LDP, lane);
            h16v bv = load_b_nm(&ldsV[no * 16][ks * 32], HL, lane);
            ao = wmma16(ap, bv, ao);
        }
        for (int r = 0; r < 8; r++) {
            int qpos = wv * 16 + r + 8 * g;
            int ch = head * HD + no * 16 + nlo;
            attnout[((size_t)win * WL + qpos) * C + ch] = (_Float16)ao[r];
        }
    }
}

// ---------------- kernel 7: output projection + un-window + crop ----------------
__global__ __launch_bounds__(128) void k_proj(const _Float16* __restrict__ attnout,
                                              const _Float16* __restrict__ projT,
                                              const float* __restrict__ proj_b,
                                              float* __restrict__ out) {
    int win = blockIdx.x;
    int lane = threadIdx.x & 31, wv = threadIdx.x >> 5;
    int wj = win & 15, wi = (win >> 4) & 15, b = win >> 8;
    f8v acc[16] = {};
    for (int ks = 0; ks < 8; ks++) {
        h16v a = load_a_rm(attnout + ((size_t)win * WL + wv * 16) * C + ks * 32, C, lane);
        for (int t = 0; t < 16; t++) {
            h16v bf = load_b_nm(projT + (size_t)(t * 16) * C + ks * 32, C, lane);
            acc[t] = wmma16(a, bf, acc[t]);
        }
    }
    int g = lane >> 4, nlo = lane & 15;
    for (int t = 0; t < 16; t++) {
        int n = t * 16 + nlo;
        float pb = proj_b[n];
        for (int r = 0; r < 8; r++) {
            int qpos = wv * 16 + r + 8 * g;
            int h = wi * 8 + (qpos >> 3), w = wj * 8 + (qpos & 7);
            if (h < HIM && w < WIM)
                out[(((size_t)b * HIM + h) * WIM + w) * C + n] = acc[t][r] + pb;
        }
    }
}

// ---------------- launch ----------------
extern "C" void kernel_launch(void* const* d_in, const int* in_sizes, int n_in,
                              void* d_out, int out_size, void* d_ws, size_t ws_size,
                              hipStream_t stream) {
    const float* x           = (const float*)d_in[0];
    const float* dw_kernel   = (const float*)d_in[1];
    const float* ln_gamma    = (const float*)d_in[2];
    const float* ln_beta     = (const float*)d_in[3];
    const float* w_qkv       = (const float*)d_in[4];
    const float* q_bias      = (const float*)d_in[5];
    const float* v_bias      = (const float*)d_in[6];
    const float* scale_logit = (const float*)d_in[7];
    const float* cpb_w0      = (const float*)d_in[8];
    const float* cpb_b0      = (const float*)d_in[9];
    const float* cpb_w1      = (const float*)d_in[10];
    const float* proj_w      = (const float*)d_in[11];
    const float* proj_b      = (const float*)d_in[12];
    float* out = (float*)d_out;

    char* ws = (char*)d_ws;
    size_t off = 0;
    auto alloc = [&](size_t bytes) -> void* {
        void* p = ws + off;
        off += (bytes + 255) & ~(size_t)255;
        return p;
    };
    _Float16* y       = (_Float16*)alloc((size_t)NPIX * C * 2);          // 16.8 MB
    _Float16* qkv     = (_Float16*)alloc((size_t)NPIX * C3 * 2);         // 50.3 MB
    _Float16* qL2     = (_Float16*)alloc((size_t)NWIN * HEADS * WL * HD * 2);
    _Float16* kL2     = (_Float16*)alloc((size_t)NWIN * HEADS * HL * HD * 2);
    _Float16* vT      = (_Float16*)alloc((size_t)NWIN * HEADS * HD * HL * 2);
    _Float16* attnout = (_Float16*)alloc((size_t)NWIN * WL * C * 2);     // 16.8 MB
    _Float16* wqkvT   = (_Float16*)alloc((size_t)C3 * C * 2);
    _Float16* projT   = (_Float16*)alloc((size_t)C * C * 2);
    float*    tab     = (float*)alloc((size_t)529 * 8 * 4);

    (void)in_sizes; (void)n_in; (void)out_size; (void)ws_size;

    k_prep<<<(C3 * C + C * C + 255) / 256, 256, 0, stream>>>(w_qkv, proj_w, wqkvT, projT);
    k_cpb<<<(529 * 8 + 255) / 256, 256, 0, stream>>>(cpb_w0, cpb_b0, cpb_w1, tab);
    k_conv_ln<<<NPIX, 256, 0, stream>>>(x, dw_kernel, ln_gamma, ln_beta, y);
    k_qkv<<<(NPIX / 64) * 6, 128, 0, stream>>>(y, wqkvT, q_bias, v_bias, qkv);
    k_gather<<<(NWIN * HL * HEADS) / 256, 256, 0, stream>>>(qkv, scale_logit, qL2, kL2, vT);
    k_attn<<<NWIN * HEADS, 128, 0, stream>>>(qL2, kL2, vT, tab, attnout);
    k_proj<<<NWIN, 128, 0, stream>>>(attnout, projT, proj_b, out);
}
